// SparseAttention_40948218200597
// MI455X (gfx1250) — matrix operand order repaired and checked by reading, not compiled, not run on hardware
//
#include <hip/hip_runtime.h>
#include <hip/hip_bf16.h>
#include <math.h>

typedef __attribute__((ext_vector_type(16))) _Float16 v16h;
typedef __attribute__((ext_vector_type(8)))  _Float16 v8h;
typedef __attribute__((ext_vector_type(8)))  float    v8f;

#define Bk    128
#define Tk    256
#define Ck    384
#define Hk    8
#define HDk   48
#define WINk  64
#define C3    1152
#define Mrows (Bk*Tk)                 /* 32768 */
#define PLANE ((size_t)Bk*Hk*Tk*HDk)  /* 12,582,912 halves per q/k/v plane */

static __device__ __forceinline__ v8f wmma16(v16h a, v16h b, v8f c) {
  // (neg_a, A, neg_b, B, c_mod, C, reuse_a, reuse_b)
  return __builtin_amdgcn_wmma_f32_16x16x32_f16(false, a, false, b, (short)0, c,
                                                false, false);
}

// Async 16-byte global->LDS copy (per active lane). VDST = LDS byte address,
// VADDR = 64-bit global address, tracked by ASYNCcnt (ISA 15.18.3 op 98).
static __device__ __forceinline__ void async_b128(unsigned ldsAddr, const void* g) {
  asm volatile("global_load_async_to_lds_b128 %0, %1, off"
               :: "v"(ldsAddr), "v"((unsigned long long)(uintptr_t)g)
               : "memory");
}
static __device__ __forceinline__ void wait_async0() {
  asm volatile("s_wait_asynccnt 0x0" ::: "memory");
}
static __device__ __forceinline__ unsigned lds_addr(const void* p) {
  // generic LDS pointer: low 32 bits are the LDS byte offset (aperture in high)
  return (unsigned)(uintptr_t)p;
}

// ---------------------------------------------------------------------------
// Kernel 1: qkv = x @ W_qkv + b_qkv  -> scattered into f16 head-major planes.
//   256 threads (8 waves), 128x64 output tile, K staged 32 at a time.
//   A/B tiles staged as raw f32 via async-to-LDS; f32->f16 at fragment build.
// ---------------------------------------------------------------------------
__global__ __launch_bounds__(256) void qkv_gemm_kernel(
    const float* __restrict__ x, const float* __restrict__ Wqkv,
    const float* __restrict__ bqkv, _Float16* __restrict__ qkvws) {
  __shared__ float Asf[128][36];   // 128 x 32 f32 (+4 pad: rows 144B, 16B-mult)
  __shared__ float Bsf[32][64];    // 32  x 64 f32 (rows 256B)

  const int tid    = threadIdx.x;
  const int lane   = tid & 31;
  const int laneLo = lane & 15;
  const int laneHi = lane >> 4;
  const int wave   = tid >> 5;
  const int wm     = wave & 3;
  const int wn     = wave >> 2;
  const int rowBase = blockIdx.x * 128;
  const int colBase = blockIdx.y * 64;

  v8f acc[2][2] = {};

  for (int kb = 0; kb < Ck; kb += 32) {
    // A tile: 128x32 f32 = 1024 x (4-float chunks), 4 per thread
    for (int it = 0; it < 4; ++it) {
      int e = it * 256 + tid;
      int r = e >> 3, c4 = (e & 7) * 4;
      async_b128(lds_addr(&Asf[r][c4]),
                 x + (size_t)(rowBase + r) * Ck + kb + c4);
    }
    // B tile: 32x64 f32 = 512 chunks, 2 per thread
    for (int it = 0; it < 2; ++it) {
      int e = it * 256 + tid;
      int r = e >> 4, c4 = (e & 15) * 4;
      async_b128(lds_addr(&Bsf[r][c4]),
                 Wqkv + (size_t)(kb + r) * C3 + colBase + c4);
    }
    wait_async0();
    __syncthreads();

    v16h af[2], bf[2];
    for (int mi = 0; mi < 2; ++mi) {
      int row = wm * 32 + mi * 16 + laneLo;
      for (int j = 0; j < 8; ++j) {
        af[mi][j]     = (_Float16)Asf[row][8 * laneHi + j];
        af[mi][8 + j] = (_Float16)Asf[row][16 + 8 * laneHi + j];
      }
    }
    for (int ni = 0; ni < 2; ++ni) {
      int col = wn * 32 + ni * 16 + laneLo;
      for (int j = 0; j < 16; ++j) bf[ni][j] = (_Float16)Bsf[(j & 7) + 8 * laneHi + 16 * (j >> 3)][col];
    }
    for (int mi = 0; mi < 2; ++mi)
      for (int ni = 0; ni < 2; ++ni)
        acc[mi][ni] = wmma16(af[mi], bf[ni], acc[mi][ni]);
    __syncthreads();
  }

  for (int mi = 0; mi < 2; ++mi) {
    for (int ni = 0; ni < 2; ++ni) {
      int ng = colBase + wn * 32 + ni * 16 + laneLo;
      int which = ng / Ck;
      int rem   = ng - which * Ck;
      int h     = rem / HDk;
      int d     = rem - h * HDk;
      float bias = bqkv[ng];
      for (int r = 0; r < 8; ++r) {
        int mg = rowBase + wm * 32 + mi * 16 + r + 8 * laneHi;
        int bb = mg >> 8;
        int tt = mg & 255;
        float v = acc[mi][ni][r] + bias;
        qkvws[(size_t)which * PLANE +
              (((size_t)bb * Hk + h) * Tk + tt) * HDk + d] = (_Float16)v;
      }
    }
  }
}

// ---------------------------------------------------------------------------
// Kernel 2: windowed-causal flash attention.
//   Block = 8 waves sharing one (b, h) and 128 consecutive queries.
//   Block-wide K/V footprint = 192 keys x 48 -> async-staged to LDS once;
//   per-wave: S = QK^T (2 WMMA x 5 tiles), masked softmax in-register,
//   P relaid via LDS, O = P@V (9 WMMA), gate fused at the f16 store.
// ---------------------------------------------------------------------------
__global__ __launch_bounds__(256) void attn_kernel(
    const _Float16* __restrict__ qkvws, const float* __restrict__ gate,
    _Float16* __restrict__ attnws) {
  __shared__ _Float16 Kl[192][48];     // 18 KB
  __shared__ _Float16 Vl[192][48];     // 18 KB
  __shared__ _Float16 Ps[8][16][96];   // 24 KB

  const int tid    = threadIdx.x;
  const int lane   = tid & 31;
  const int laneLo = lane & 15;
  const int laneHi = lane >> 4;
  const int wave   = tid >> 5;

  const int blk   = blockIdx.x;
  const int qblk0 = 128 * (blk & 1);          // first query of this block
  const int h     = (blk >> 1) & 7;
  const int b     = blk >> 4;
  const int kb0   = qblk0 - WINk;             // first (possibly negative) key
  const int qb    = qblk0 + wave * 16;        // this wave's query tile

  const _Float16* qPlane = qkvws + ((size_t)(b * Hk + h) * Tk) * HDk;
  const _Float16* kPlane = qPlane + PLANE;
  const _Float16* vPlane = qPlane + 2 * PLANE;

  // ---- stage K/V block-wide: 192 rows x 6 chunks of 8 halves = 1152 each ----
  for (int it = 0; it < 5; ++it) {
    int e = it * 256 + tid;
    if (e < 1152) {
      int row  = e / 6;
      int colc = (e - row * 6) * 8;
      int key  = kb0 + row;
      if (key >= 0) {
        async_b128(lds_addr(&Kl[row][colc]), kPlane + (size_t)key * HDk + colc);
        async_b128(lds_addr(&Vl[row][colc]), vPlane + (size_t)key * HDk + colc);
      } else {
        v8h z = {};
        *(v8h*)&Kl[row][colc] = z;   // keep masked region finite (0 * NaN trap)
        *(v8h*)&Vl[row][colc] = z;
      }
    }
  }
  wait_async0();
  __syncthreads();

  // ---- Q fragments (A-layout), HD=48 padded to 64, direct from global ----
  v16h qf[2];
  {
    const _Float16* qr = qPlane + (size_t)(qb + laneLo) * HDk;
    for (int j = 0; j < 8; ++j) {
      qf[0][j]     = qr[8 * laneHi + j];
      qf[0][8 + j] = qr[16 + 8 * laneHi + j];
      qf[1][j]     = qr[32 + 8 * laneHi + j];
      qf[1][8 + j] = (_Float16)0.0f;
    }
  }

  // ---- S = scale * Q K^T over 5 key tiles (keys qb-64 .. qb+15) ----
  float sAll[5][8];
  const float scale = 0.14433756729740643f;    // 1/sqrt(48)
  for (int t = 0; t < 5; ++t) {
    int kb    = qb - WINk + 16 * t;
    int krow  = (kb - kb0) + laneLo;           // 0..191, always in range
    v16h kf0, kf1;
    for (int j = 0; j < 16; ++j) {
      kf0[j] = Kl[krow][(j & 7) + 8 * laneHi + 16 * (j >> 3)];
      int d = 32 + (j & 7) + 8 * laneHi + 16 * (j >> 3);
      kf1[j] = (d < HDk) ? Kl[krow][d] : (_Float16)0.0f;
    }
    v8f s = {};
    s = wmma16(qf[0], kf0, s);
    s = wmma16(qf[1], kf1, s);
    for (int r = 0; r < 8; ++r) {
      int q   = qb + r + 8 * laneHi;
      int key = kb + laneLo;
      float sv = s[r] * scale;
      sAll[t][r] = (key >= 0 && q >= key && (q - key) <= WINk)
                     ? sv : -__builtin_inff();
    }
  }

  // ---- row softmax across 80 keys ----
  float inv[8];
  for (int r = 0; r < 8; ++r) {
    float m = sAll[0][r];
    for (int t = 1; t < 5; ++t) m = fmaxf(m, sAll[t][r]);
    for (int off = 1; off < 16; off <<= 1) m = fmaxf(m, __shfl_xor(m, off, 32));
    float su = 0.f;
    for (int t = 0; t < 5; ++t) {
      float p = __expf(sAll[t][r] - m);        // exp(-inf)=0 handles masks
      sAll[t][r] = p;
      su += p;
    }
    for (int off = 1; off < 16; off <<= 1) su += __shfl_xor(su, off, 32);
    inv[r] = 1.0f / su;
  }

  // ---- stage normalized P to LDS (C-layout -> A-layout relay), pad to 96 ----
  for (int t = 0; t < 6; ++t) {
    for (int r = 0; r < 8; ++r) {
      int m = r + 8 * laneHi;
      float p = (t < 5) ? sAll[t][r] * inv[r] : 0.0f;
      Ps[wave][m][t * 16 + laneLo] = (_Float16)p;
    }
  }
  __syncthreads();

  // ---- O = P @ V : 3 K-chunks of 32 keys x 3 d-tiles of 16 ----
  v8f oacc[3] = {};
  for (int c = 0; c < 3; ++c) {
    v16h pa;
    for (int j = 0; j < 16; ++j) {
      int keyoff = 32 * c + (j & 7) + 8 * laneHi + 16 * (j >> 3);
      pa[j] = Ps[wave][laneLo][keyoff];
    }
    for (int nt = 0; nt < 3; ++nt) {
      v16h vb;
      int d = nt * 16 + laneLo;
      for (int j = 0; j < 16; ++j) {
        int kidx = 32 * c + (j & 7) + 8 * laneHi + 16 * (j >> 3);       // 0..95
        // LDS row = wave*16 + kidx; key<0 rows are zero, P there is 0 anyway
        vb[j] = (kidx < 80) ? Vl[wave * 16 + kidx][d] : (_Float16)0.0f;
      }
      oacc[nt] = wmma16(pa, vb, oacc[nt]);
    }
  }

  // ---- gated f16 store into [B*T, C] row-major for the proj GEMM ----
  for (int nt = 0; nt < 3; ++nt) {
    int d  = nt * 16 + laneLo;
    int ch = h * HDk + d;
    float g = gate[ch];
    for (int r = 0; r < 8; ++r) {
      int q = qb + r + 8 * laneHi;
      attnws[((size_t)(b * Tk + q)) * Ck + ch] = (_Float16)(oacc[nt][r] * g);
    }
  }
}

// ---------------------------------------------------------------------------
// Kernel 3: y = attn_gated(f16) @ W_proj + b_proj -> f32 out.
//   A tile async-staged as f16; B tile async-staged as raw f32.
// ---------------------------------------------------------------------------
__global__ __launch_bounds__(256) void proj_gemm_kernel(
    const _Float16* __restrict__ attnws, const float* __restrict__ Wp,
    const float* __restrict__ bp, float* __restrict__ out) {
  __shared__ _Float16 As[128][40];   // 128 x 32 f16 (+8 pad: rows 80B)
  __shared__ float    Bsf[32][64];   // 32 x 64 f32

  const int tid    = threadIdx.x;
  const int lane   = tid & 31;
  const int laneLo = lane & 15;
  const int laneHi = lane >> 4;
  const int wave   = tid >> 5;
  const int wm     = wave & 3;
  const int wn     = wave >> 2;
  const int rowBase = blockIdx.x * 128;
  const int colBase = blockIdx.y * 64;

  v8f acc[2][2] = {};

  for (int kb = 0; kb < Ck; kb += 32) {
    // A tile: 128x32 halves = 512 x (8-half chunks), 2 per thread
    for (int it = 0; it < 2; ++it) {
      int e = it * 256 + tid;
      int r = e >> 2, c8 = (e & 3) * 8;
      async_b128(lds_addr(&As[r][c8]),
                 attnws + (size_t)(rowBase + r) * Ck + kb + c8);
    }
    // B tile: 32x64 f32 = 512 chunks, 2 per thread
    for (int it = 0; it < 2; ++it) {
      int e = it * 256 + tid;
      int r = e >> 4, c4 = (e & 15) * 4;
      async_b128(lds_addr(&Bsf[r][c4]),
                 Wp + (size_t)(kb + r) * Ck + colBase + c4);
    }
    wait_async0();
    __syncthreads();

    v16h af[2], bf[2];
    for (int mi = 0; mi < 2; ++mi) {
      int row = wm * 32 + mi * 16 + laneLo;
      for (int j = 0; j < 8; ++j) {
        af[mi][j]     = As[row][8 * laneHi + j];
        af[mi][8 + j] = As[row][16 + 8 * laneHi + j];
      }
    }
    for (int ni = 0; ni < 2; ++ni) {
      int col = wn * 32 + ni * 16 + laneLo;
      for (int j = 0; j < 16; ++j) bf[ni][j] = (_Float16)Bsf[(j & 7) + 8 * laneHi + 16 * (j >> 3)][col];
    }
    for (int mi = 0; mi < 2; ++mi)
      for (int ni = 0; ni < 2; ++ni)
        acc[mi][ni] = wmma16(af[mi], bf[ni], acc[mi][ni]);
    __syncthreads();
  }

  for (int mi = 0; mi < 2; ++mi) {
    for (int ni = 0; ni < 2; ++ni) {
      int ng = colBase + wn * 32 + ni * 16 + laneLo;
      float bias = bp[ng];
      for (int r = 0; r < 8; ++r) {
        int mg = rowBase + wm * 32 + mi * 16 + r + 8 * laneHi;
        out[(size_t)mg * Ck + ng] = acc[mi][ni][r] + bias;
      }
    }
  }
}

// ---------------------------------------------------------------------------
extern "C" void kernel_launch(void* const* d_in, const int* in_sizes, int n_in,
                              void* d_out, int out_size, void* d_ws,
                              size_t ws_size, hipStream_t stream) {
  const float* x     = (const float*)d_in[0];
  const float* Wqkv  = (const float*)d_in[1];
  const float* bqkv  = (const float*)d_in[2];
  const float* Wproj = (const float*)d_in[3];
  const float* bproj = (const float*)d_in[4];
  const float* gate  = (const float*)d_in[5];
  float* out = (float*)d_out;

  _Float16* qkvws  = (_Float16*)d_ws;              // 3 * PLANE halves
  _Float16* attnws = qkvws + 3 * PLANE;            // Mrows*Ck halves

  dim3 g1(Mrows / 128, C3 / 64);                   // 256 x 18
  qkv_gemm_kernel<<<g1, 256, 0, stream>>>(x, Wqkv, bqkv, qkvws);

  int nBlocks = (Bk * Hk * Tk) / 16 / 8;           // 2048
  attn_kernel<<<nBlocks, 256, 0, stream>>>(qkvws, gate, attnws);

  dim3 g3(Mrows / 128, Ck / 64);                   // 256 x 6
  proj_gemm_kernel<<<g3, 256, 0, stream>>>(attnws, Wproj, bproj, out);
}
